// Discriminator_25022479466569
// MI455X (gfx1250) — matrix operand order repaired and checked by reading, not compiled, not run on hardware
//
#include <hip/hip_runtime.h>
#include <hip/hip_bf16.h>
#include <math.h>

// ---------------------------------------------------------------------------
// Discriminator forward, restructured after dead-code elimination:
//   * attention branch is dead (del attn) -> skipped entirely
//   * only node 0 and node N-1 per batch feed the live path
//   * all matmuls run as v_wmma_f32_16x16x32_f16 (f16 in, f32 accumulate)
//   * weights stored transposed so B fragments are two contiguous b128 loads
//   * GEMMs templated on <M,N,K> so the K loop fully unrolls
// ---------------------------------------------------------------------------

typedef _Float16 v8h  __attribute__((ext_vector_type(8)));
typedef _Float16 v16h __attribute__((ext_vector_type(16)));
typedef float    v8f  __attribute__((ext_vector_type(8)));

#define B_SZ   256
#define N_SZ   200
#define M_SZ   200
#define KPAD1  768   // 752 padded to multiple of 32

// ---------------- weight convert: f32 (K x N) -> f16 transposed (N x Kpad) --
__global__ void transpose_pad_f16(const float* __restrict__ src,
                                  _Float16* __restrict__ dst,
                                  int Ksrc, int Kpad, int Ncols) {
  int idx = blockIdx.x * blockDim.x + threadIdx.x;
  int total = Ncols * Kpad;
  if (idx >= total) return;
  int n = idx / Kpad;
  int k = idx - n * Kpad;
  dst[idx] = (k < Ksrc) ? (_Float16)src[(size_t)k * Ncols + n] : (_Float16)0.0f;
}

// ---------------- gather node vectors ---------------------------------------
// Row 2b   = node_vec(b, N-1)  ("last")
// Row 2b+1 = node_vec(b, 0)    ("first")
// Layout per row (752 live cols, padded to 768):
//   [0..7]    pe[trees[b,j,0]]
//   [8..15]   pe[trees[b,j,1]]
//   [16..239] emb[trees[b,j,2..15]] (14 x 16)
//   [240..751] lstm_out_list[b, trees[b,j,16], 0, :]
__global__ void gather_nodes(const float* __restrict__ lstm,
                             const int*   __restrict__ trees,
                             const float* __restrict__ emb,
                             _Float16*    __restrict__ Xh) {
  int row = blockIdx.x;                 // 0..511
  int b   = row >> 1;
  int j   = (row & 1) ? 0 : (N_SZ - 1);
  const int* t = trees + ((size_t)b * N_SZ + j) * 17;
  int t0  = t[0];
  int t1  = t[1];
  int ptr = t[16];
  _Float16* dst = Xh + (size_t)row * KPAD1;
  const float neg_ln1e4_over_d = -logf(10000.0f) / 8.0f;
  for (int c = threadIdx.x; c < KPAD1; c += blockDim.x) {
    float v;
    if (c < 16) {
      int p  = (c < 8) ? t0 : t1;
      int cc = c & 7;
      int i2 = (cc >> 1) * 2;                       // 0,2,4,6
      float div = expf((float)i2 * neg_ln1e4_over_d);
      float ang = (float)p * div;
      v = (cc & 1) ? cosf(ang) : sinf(ang);
    } else if (c < 240) {
      int e = (c - 16) >> 4;                        // 0..13
      int d = (c - 16) & 15;
      v = emb[(size_t)t[2 + e] * 16 + d];
    } else if (c < 752) {
      v = lstm[((size_t)b * M_SZ + ptr) * 512 + (c - 240)];
    } else {
      v = 0.0f;                                     // K padding
    }
    dst[c] = (_Float16)v;
  }
}

// ---------------- WMMA fragment loaders (per CDNA5 ISA layouts) -------------
__device__ inline v16h load_a_frag(const _Float16* __restrict__ A, int lda,
                                   int row, int k0, int lane) {
  // 16-bit A 16x32: lanes 0-15 -> K = e (+16 for e>=8); lanes 16-31 -> +8
  int base = (lane & 16) ? 8 : 0;
  const v8h* p0 = (const v8h*)(A + row * lda + k0 + base);
  const v8h* p1 = (const v8h*)(A + row * lda + k0 + 16 + base);
  v8h lo = *p0;
  v8h hi = *p1;
  v16h r;
#pragma unroll
  for (int i = 0; i < 8; ++i) { r[i] = lo[i]; r[i + 8] = hi[i]; }
  return r;
}

__device__ inline v16h load_bt_frag(const _Float16* __restrict__ Bt, int ldb,
                                    int k0, int col, int lane) {
  // 16-bit B 32x16: lanes 0-15 hold K = e (+16 for e>=8); lanes 16-31 -> +8.
  // Bt is N x Kpad (transposed), so each 8 K-values are contiguous.
  int kb = k0 + ((lane & 16) ? 8 : 0);
  const v8h* p = (const v8h*)(Bt + col * ldb + kb);
  v8h lo = p[0];
  v8h hi = p[2];
  v16h r;
#pragma unroll
  for (int i = 0; i < 8; ++i) { r[i] = lo[i]; r[i + 8] = hi[i]; }
  return r;
}

// ---------------- WMMA GEMM: C = relu(A @ B + bias), B given transposed -----
// A: (M,K) f16 row-major, Bt: (N,K) f16 row-major, C: (M,N) f16.
// One wave (32 lanes) per 16x16 output tile; 8 waves per block.
template <int M, int N, int K>
__global__ void wmma_gemm_bias_relu(const _Float16* __restrict__ A,
                                    const _Float16* __restrict__ Bt,
                                    const float*    __restrict__ bias,
                                    _Float16*       __restrict__ C) {
  constexpr int tiles_n = N >> 4;
  constexpr int tiles   = (M >> 4) * tiles_n;
  int wave = threadIdx.x >> 5;
  int lane = threadIdx.x & 31;
  int tile = blockIdx.x * 8 + wave;
  if (tile >= tiles) return;                 // whole-wave exit: EXEC stays uniform
  int tm = tile / tiles_n;
  int tn = tile - tm * tiles_n;

  int r16  = lane & 15;
  int hi16 = (lane >> 4) & 1;
  int arow = tm * 16 + r16;                  // this lane's A row
  int bcol = tn * 16 + r16;                  // this lane's B/C column

  // pull this wave's A row / Bt column toward the WGP before fragment loads
  __builtin_prefetch(A + arow * K, 0, 3);
  __builtin_prefetch(Bt + bcol * K, 0, 3);

  v8f acc = {};
#pragma unroll
  for (int k0 = 0; k0 < K; k0 += 32) {
    v16h a = load_a_frag(A, K, arow, k0, lane);
    v16h b = load_bt_frag(Bt, K, k0, bcol, lane);
    acc = __builtin_amdgcn_wmma_f32_16x16x32_f16(
        /*neg_a=*/false, a, /*neg_b=*/false, b,
        /*c_mod=*/(short)0, acc, /*reuse_a=*/false, /*reuse_b=*/false);
  }

  float bv = bias[bcol];
  int cbase = (tm * 16 + hi16 * 8) * N + bcol;   // 32-bit addressing
#pragma unroll
  for (int r = 0; r < 8; ++r) {
    float v = fmaxf(acc[r] + bv, 0.0f);
    C[cbase + r * N] = (_Float16)v;
  }
}

// ---------------- rearrange Y2 (512x32) -> H (256x192) ----------------------
// H[b] = [ h(last) | h(first) x5 ]
__global__ void build_h(const _Float16* __restrict__ Y2,
                        _Float16* __restrict__ H) {
  int idx = blockIdx.x * blockDim.x + threadIdx.x;
  if (idx >= B_SZ * 192) return;
  int b = idx / 192;
  int c = idx - b * 192;
  int seg = c >> 5;
  int d   = c & 31;
  int src = (seg == 0) ? (2 * b) : (2 * b + 1);
  H[idx] = Y2[src * 32 + d];
}

// ---------------- final: out[b] = F2[b,:] . ts_w + ts_b ---------------------
__global__ void final_ts(const _Float16* __restrict__ F2,
                         const float* __restrict__ ts_w,
                         const float* __restrict__ ts_b,
                         float* __restrict__ out) {
  int b = blockIdx.x * blockDim.x + threadIdx.x;
  if (b >= B_SZ) return;
  float s = ts_b[0];
#pragma unroll
  for (int k = 0; k < 32; ++k) s += (float)F2[b * 32 + k] * ts_w[k];
  out[b] = s;
}

// ---------------------------------------------------------------------------
extern "C" void kernel_launch(void* const* d_in, const int* in_sizes, int n_in,
                              void* d_out, int out_size, void* d_ws, size_t ws_size,
                              hipStream_t stream) {
  const float* lstm    = (const float*)d_in[0];
  const int*   trees   = (const int*)  d_in[2];
  const float* emb     = (const float*)d_in[4];
  const float* node_w1 = (const float*)d_in[9];
  const float* node_b1 = (const float*)d_in[10];
  const float* node_w2 = (const float*)d_in[11];
  const float* node_b2 = (const float*)d_in[12];
  const float* ta_w1   = (const float*)d_in[13];
  const float* ta_b1   = (const float*)d_in[14];
  const float* ta_w2   = (const float*)d_in[15];
  const float* ta_b2   = (const float*)d_in[16];
  const float* ff_w1   = (const float*)d_in[17];
  const float* ff_b1   = (const float*)d_in[18];
  const float* ff_w2   = (const float*)d_in[19];
  const float* ff_b2   = (const float*)d_in[20];
  const float* ts_w    = (const float*)d_in[21];
  const float* ts_b    = (const float*)d_in[22];
  float* out = (float*)d_out;

  char* ws = (char*)d_ws;
  _Float16* Xh   = (_Float16*)(ws + 0);        // 512 x 768
  _Float16* W1t  = (_Float16*)(ws + 786432);   // 128 x 768 (transposed)
  _Float16* Y1h  = (_Float16*)(ws + 983040);   // 512 x 128
  _Float16* W2t  = (_Float16*)(ws + 1114112);  // 32 x 128 (transposed)
  _Float16* Y2h  = (_Float16*)(ws + 1122304);  // 512 x 32
  _Float16* Hh   = (_Float16*)(ws + 1155072);  // 256 x 192
  _Float16* TA1t = (_Float16*)(ws + 1253376);  // 128 x 192 (transposed)
  _Float16* T1h  = (_Float16*)(ws + 1302528);  // 256 x 128
  _Float16* TA2t = (_Float16*)(ws + 1368064);  // 32 x 128 (transposed)
  _Float16* T2h  = (_Float16*)(ws + 1376256);  // 256 x 32
  _Float16* FF1t = (_Float16*)(ws + 1392640);  // 64 x 32 (transposed)
  _Float16* F1h  = (_Float16*)(ws + 1396736);  // 256 x 64
  _Float16* FF2t = (_Float16*)(ws + 1429504);  // 32 x 64 (transposed)
  _Float16* F2h  = (_Float16*)(ws + 1433600);  // 256 x 32

  // weights -> f16, transposed to (N x Kpad); node_w1 K-padded 752 -> 768
  auto cvt = [&](const float* s, _Float16* d, int Ksrc, int Kpad, int Ncols) {
    int tot = Ncols * Kpad;
    transpose_pad_f16<<<(tot + 255) / 256, 256, 0, stream>>>(s, d, Ksrc, Kpad, Ncols);
  };
  cvt(node_w1, W1t, 752, KPAD1, 128);
  cvt(node_w2, W2t, 128, 128, 32);
  cvt(ta_w1,  TA1t, 192, 192, 128);
  cvt(ta_w2,  TA2t, 128, 128, 32);
  cvt(ff_w1,  FF1t, 32,  32,  64);
  cvt(ff_w2,  FF2t, 64,  64,  32);

  // gather live node vectors (512 rows)
  gather_nodes<<<512, 256, 0, stream>>>(lstm, trees, emb, Xh);

  // GEMM chain (tiles = (M/16)*(N/16), 8 waves per 256-thread block)
  wmma_gemm_bias_relu<512, 128, KPAD1><<<32, 256, 0, stream>>>(Xh, W1t, node_b1, Y1h);
  wmma_gemm_bias_relu<512, 32, 128><<<8, 256, 0, stream>>>(Y1h, W2t, node_b2, Y2h);
  build_h<<<(B_SZ * 192 + 255) / 256, 256, 0, stream>>>(Y2h, Hh);
  wmma_gemm_bias_relu<256, 128, 192><<<16, 256, 0, stream>>>(Hh, TA1t, ta_b1, T1h);
  wmma_gemm_bias_relu<256, 32, 128><<<4, 256, 0, stream>>>(T1h, TA2t, ta_b2, T2h);
  wmma_gemm_bias_relu<256, 64, 32><<<8, 256, 0, stream>>>(T2h, FF1t, ff_b1, F1h);
  wmma_gemm_bias_relu<256, 32, 64><<<4, 256, 0, stream>>>(F1h, FF2t, ff_b2, F2h);
  final_ts<<<1, 256, 0, stream>>>(F2h, ts_w, ts_b, out);
}
